// QLSTM_65481071399935
// MI455X (gfx1250) — matrix operand order repaired and checked by reading, not compiled, not run on hardware
//
#include <hip/hip_runtime.h>

#define S_LEN 512
#define BATCH 64
#define DIM   1024
#define HID   1024
#define G4    4096   // 4*HID

typedef __attribute__((ext_vector_type(16))) __bf16 v16bf;
typedef __attribute__((ext_vector_type(8)))  float  v8f;

union FragU { v16bf v; struct { uint4 a, b; } q; };

__device__ __forceinline__ unsigned short f2bf(float f) {
  unsigned u = __float_as_uint(f);
  u += 0x7FFFu + ((u >> 16) & 1u);     // round-to-nearest-even
  return (unsigned short)(u >> 16);
}

__device__ __forceinline__ float sigm(float x) { return 1.0f / (1.0f + expf(-x)); }

// ---------------- prep kernels ----------------
__global__ void __launch_bounds__(256) k_f2bf(const float* __restrict__ src,
                                              unsigned short* __restrict__ dst, int n) {
  int i = blockIdx.x * blockDim.x + threadIdx.x;
  if (i < n) dst[i] = f2bf(src[i]);
}

__global__ void __launch_bounds__(256) k_addv(const float* __restrict__ a,
                                              const float* __restrict__ b,
                                              float* __restrict__ o, int n) {
  int i = blockIdx.x * blockDim.x + threadIdx.x;
  if (i < n) o[i] = a[i] + b[i];
}

__global__ void __launch_bounds__(256) k_zero(unsigned int* p, int n) {
  int i = blockIdx.x * blockDim.x + threadIdx.x;
  if (i < n) p[i] = 0u;
}

// ---------------- WMMA fragment loads ----------------
// A (16x32 bf16, row-major source, row stride = K): lane half h picks
// K in {kb+8h .. kb+8h+7} U {kb+16+8h .. kb+16+8h+7}
__device__ __forceinline__ v16bf load_a(const unsigned short* __restrict__ rowk, int half) {
  FragU u;
  u.q.a = *(const uint4*)(rowk + half * 8);
  u.q.b = *(const uint4*)(rowk + 16 + half * 8);
  return u.v;
}
// B (32x16 bf16). Source is W row-major [N][K] == B^T, so column n of B is
// row n of W, K-contiguous. Lane half h picks
// K in {kb+8h .. kb+8h+7} U {kb+16+8h .. kb+16+8h+7}
__device__ __forceinline__ v16bf load_b(const unsigned short* __restrict__ rowk, int half) {
  FragU u;
  u.q.a = *(const uint4*)(rowk + half * 8);
  u.q.b = *(const uint4*)(rowk + 16 + half * 8);
  return u.v;
}

// Z[64][4096] = [Ax | Ah] @ [Wi | Wh]^T + bias   (bf16 operands, f32 accum)
// grid = 256 blocks (one per 16-col strip), 128 threads = 4 waves,
// wave w owns m-tile w (rows 16w..16w+15). B fragments shared via L1.
__global__ void __launch_bounds__(128) k_gemm_gates(
    const unsigned short* __restrict__ Ax,   // [64][1024] bf16
    const unsigned short* __restrict__ Ah,   // [64][1024] bf16
    const unsigned short* __restrict__ Wi,   // [4096][1024] bf16 (row-major = B^T)
    const unsigned short* __restrict__ Wh,   // [4096][1024] bf16
    const float*          __restrict__ bias, // [4096]
    float*                __restrict__ Z)    // [64][4096]
{
  const int lane = threadIdx.x & 31;
  const int wave = threadIdx.x >> 5;
  const int half = lane >> 4;
  const int r    = lane & 15;
  const int nbase = blockIdx.x * 16;
  const int m     = wave * 16 + r;

  const unsigned short* arow_x = Ax + (size_t)m * DIM;
  const unsigned short* arow_h = Ah + (size_t)m * HID;
  const unsigned short* brow_i = Wi + (size_t)(nbase + r) * DIM;
  const unsigned short* brow_h = Wh + (size_t)(nbase + r) * HID;

  v8f acc = {};
#pragma unroll 4
  for (int kb = 0; kb < DIM; kb += 32) {
    v16bf a = load_a(arow_x + kb, half);
    v16bf b = load_b(brow_i + kb, half);
    acc = __builtin_amdgcn_wmma_f32_16x16x32_bf16(false, a, false, b, (short)0, acc, false, false);
  }
#pragma unroll 4
  for (int kb = 0; kb < HID; kb += 32) {
    v16bf a = load_a(arow_h + kb, half);
    v16bf b = load_b(brow_h + kb, half);
    acc = __builtin_amdgcn_wmma_f32_16x16x32_bf16(false, a, false, b, (short)0, acc, false, false);
  }

  const int   n  = nbase + r;
  const float bv = bias[n];
  const int   m0 = wave * 16 + half * 8;   // C/D layout: lane half selects M block
#pragma unroll
  for (int v = 0; v < 8; ++v)
    Z[(size_t)(m0 + v) * G4 + n] = acc[v] + bv;
}

// ---------------- cell + layernorm (layer 0) ----------------
// one block per batch row; 256 threads x 4 elements
__global__ void __launch_bounds__(256) k_cell_ln0(
    const float* __restrict__ Z, float* __restrict__ C,
    float* __restrict__ Hf, unsigned short* __restrict__ Hb,
    const float* __restrict__ gamma, const float* __restrict__ beta)
{
  const int b = blockIdx.x, tid = threadIdx.x;
  __shared__ float s1[256], s2[256];
  const float* zb = Z + (size_t)b * G4;
  float* cb = C + (size_t)b * HID;
  float hr[4], ls = 0.f, lq = 0.f;
#pragma unroll
  for (int k = 0; k < 4; ++k) {
    int j = tid + k * 256;
    float c2 = sigm(zb[HID + j]) * cb[j] + sigm(zb[j]) * tanhf(zb[2 * HID + j]);
    cb[j] = c2;
    float h = sigm(zb[3 * HID + j]) * tanhf(c2);
    hr[k] = h; ls += h; lq += h * h;
  }
  s1[tid] = ls; s2[tid] = lq; __syncthreads();
  for (int off = 128; off > 0; off >>= 1) {
    if (tid < off) { s1[tid] += s1[tid + off]; s2[tid] += s2[tid + off]; }
    __syncthreads();
  }
  const float mean = s1[0] * (1.f / (float)HID);
  const float var  = s2[0] * (1.f / (float)HID) - mean * mean;
  const float rs   = rsqrtf(var + 1e-5f);
#pragma unroll
  for (int k = 0; k < 4; ++k) {
    int j = tid + k * 256;
    float hn = (hr[k] - mean) * rs * gamma[j] + beta[j];
    Hf[(size_t)b * HID + j] = hn;
    Hb[(size_t)b * HID + j] = f2bf(hn);
  }
}

// ---------------- cell + layernorm + residual + output (layer 1) ----------------
__global__ void __launch_bounds__(256) k_cell_ln1(
    const float* __restrict__ Z, float* __restrict__ C,
    const float* __restrict__ H0f, float* __restrict__ H1f,
    unsigned short* __restrict__ Hb,
    const float* __restrict__ gamma, const float* __restrict__ beta,
    float* __restrict__ Out)
{
  const int b = blockIdx.x, tid = threadIdx.x;
  __shared__ float s1[256], s2[256];
  const float* zb = Z + (size_t)b * G4;
  float* cb = C + (size_t)b * HID;
  float hr[4], ls = 0.f, lq = 0.f;
#pragma unroll
  for (int k = 0; k < 4; ++k) {
    int j = tid + k * 256;
    float c2 = sigm(zb[HID + j]) * cb[j] + sigm(zb[j]) * tanhf(zb[2 * HID + j]);
    cb[j] = c2;
    float h = sigm(zb[3 * HID + j]) * tanhf(c2);
    hr[k] = h; ls += h; lq += h * h;
  }
  s1[tid] = ls; s2[tid] = lq; __syncthreads();
  for (int off = 128; off > 0; off >>= 1) {
    if (tid < off) { s1[tid] += s1[tid + off]; s2[tid] += s2[tid + off]; }
    __syncthreads();
  }
  const float mean = s1[0] * (1.f / (float)HID);
  const float var  = s2[0] * (1.f / (float)HID) - mean * mean;
  const float rs   = rsqrtf(var + 1e-5f);
#pragma unroll
  for (int k = 0; k < 4; ++k) {
    int j = tid + k * 256;
    float hn = (hr[k] - mean) * rs * gamma[j] + beta[j] + H0f[(size_t)b * HID + j];
    H1f[(size_t)b * HID + j] = hn;
    Hb[(size_t)b * HID + j]  = f2bf(hn);
    Out[(size_t)b * HID + j] = hn;
  }
}

// final states into d_out tail: order h0, c0, h1, c1
__global__ void __launch_bounds__(256) k_final(
    const float* __restrict__ h0, const float* __restrict__ c0,
    const float* __restrict__ h1, const float* __restrict__ c1,
    float* __restrict__ out)
{
  int i = blockIdx.x * blockDim.x + threadIdx.x;
  const int N = BATCH * HID;
  if (i < N) {
    out[i]         = h0[i];
    out[N + i]     = c0[i];
    out[2 * N + i] = h1[i];
    out[3 * N + i] = c1[i];
  }
}

extern "C" void kernel_launch(void* const* d_in, const int* in_sizes, int n_in,
                              void* d_out, int out_size, void* d_ws, size_t ws_size,
                              hipStream_t stream) {
  (void)in_sizes; (void)n_in; (void)out_size; (void)ws_size;
  const float* inputs = (const float*)d_in[0];
  const float* Wih0 = (const float*)d_in[1];
  const float* Whh0 = (const float*)d_in[2];
  const float* bih0 = (const float*)d_in[3];
  const float* bhh0 = (const float*)d_in[4];
  const float* g0   = (const float*)d_in[5];
  const float* be0  = (const float*)d_in[6];
  const float* Wih1 = (const float*)d_in[7];
  const float* Whh1 = (const float*)d_in[8];
  const float* bih1 = (const float*)d_in[9];
  const float* bhh1 = (const float*)d_in[10];
  const float* g1   = (const float*)d_in[11];
  const float* be1  = (const float*)d_in[12];
  float* out = (float*)d_out;

  // -------- workspace carve --------
  char* base = (char*)d_ws;
  size_t off = 0;
  auto carve = [&](size_t bytes) -> char* {
    char* p = base + off;
    off = (off + bytes + 255) & ~(size_t)255;
    return p;
  };
  const size_t WBYTES = (size_t)G4 * DIM * sizeof(unsigned short);       // 8 MB each
  unsigned short* Wi0b = (unsigned short*)carve(WBYTES);
  unsigned short* Wh0b = (unsigned short*)carve(WBYTES);
  unsigned short* Wi1b = (unsigned short*)carve(WBYTES);
  unsigned short* Wh1b = (unsigned short*)carve(WBYTES);
  unsigned short* Xb   = (unsigned short*)carve((size_t)S_LEN * BATCH * DIM * sizeof(unsigned short)); // 64 MB
  float* bias0 = (float*)carve(G4 * sizeof(float));
  float* bias1 = (float*)carve(G4 * sizeof(float));
  float* z0    = (float*)carve((size_t)BATCH * G4 * sizeof(float));
  float* z1    = (float*)carve((size_t)BATCH * G4 * sizeof(float));
  float* c0    = (float*)carve((size_t)BATCH * HID * sizeof(float));
  float* c1    = (float*)carve((size_t)BATCH * HID * sizeof(float));
  float* h0f   = (float*)carve((size_t)BATCH * HID * sizeof(float));
  float* h1f   = (float*)carve((size_t)BATCH * HID * sizeof(float));
  unsigned short* h0b = (unsigned short*)carve((size_t)BATCH * HID * sizeof(unsigned short));
  unsigned short* h1b = (unsigned short*)carve((size_t)BATCH * HID * sizeof(unsigned short));

  // -------- one-time prep (captured in graph; deterministic) --------
  const int wn = G4 * DIM;
  k_f2bf<<<(wn + 255) / 256, 256, 0, stream>>>(Wih0, Wi0b, wn);
  k_f2bf<<<(wn + 255) / 256, 256, 0, stream>>>(Whh0, Wh0b, wn);
  k_f2bf<<<(wn + 255) / 256, 256, 0, stream>>>(Wih1, Wi1b, wn);
  k_f2bf<<<(wn + 255) / 256, 256, 0, stream>>>(Whh1, Wh1b, wn);
  const int xn = S_LEN * BATCH * DIM;
  k_f2bf<<<(xn + 255) / 256, 256, 0, stream>>>(inputs, Xb, xn);
  k_addv<<<(G4 + 255) / 256, 256, 0, stream>>>(bih0, bhh0, bias0, G4);
  k_addv<<<(G4 + 255) / 256, 256, 0, stream>>>(bih1, bhh1, bias1, G4);
  const int sn = BATCH * HID;                       // 65536 f32 words
  k_zero<<<(sn + 255) / 256, 256, 0, stream>>>((unsigned int*)c0, sn);
  k_zero<<<(sn + 255) / 256, 256, 0, stream>>>((unsigned int*)c1, sn);
  k_zero<<<(sn / 2 + 255) / 256, 256, 0, stream>>>((unsigned int*)h0b, sn / 2);
  k_zero<<<(sn / 2 + 255) / 256, 256, 0, stream>>>((unsigned int*)h1b, sn / 2);

  // -------- sequential time loop --------
  for (int t = 0; t < S_LEN; ++t) {
    const unsigned short* xt = Xb + (size_t)t * BATCH * DIM;
    k_gemm_gates<<<256, 128, 0, stream>>>(xt, h0b, Wi0b, Wh0b, bias0, z0);
    k_cell_ln0<<<BATCH, 256, 0, stream>>>(z0, c0, h0f, h0b, g0, be0);
    k_gemm_gates<<<256, 128, 0, stream>>>(h0b, h1b, Wi1b, Wh1b, bias1, z1);
    k_cell_ln1<<<BATCH, 256, 0, stream>>>(z1, c1, h0f, h1f, h1b, g1, be1,
                                          out + (size_t)t * BATCH * HID);
  }
  k_final<<<(BATCH * HID + 255) / 256, 256, 0, stream>>>(
      h0f, c0, h1f, c1, out + (size_t)S_LEN * BATCH * HID);
}
